// GNN_24893630447976
// MI455X (gfx1250) — compile-verified, matrix operand order checked by reading, not run on hardware
//
#include <hip/hip_runtime.h>

typedef __bf16          v16bf __attribute__((ext_vector_type(16)));
typedef float           v8f   __attribute__((ext_vector_type(8)));
typedef unsigned int    u32x4 __attribute__((ext_vector_type(4)));
typedef unsigned int    u32x8 __attribute__((ext_vector_type(8)));
typedef int             i32x4 __attribute__((ext_vector_type(4)));
typedef int             i32x8 __attribute__((ext_vector_type(8)));

// Problem constants
constexpr int kN  = 25000;   // nodes
constexpr int kNp = 25008;   // nodes padded to 16
constexpr int kE  = 100000;  // edges (multiple of 16)
constexpr int kC  = 64;
constexpr int kH  = 64;
constexpr int kHE = 128;
constexpr int kD  = 256;
constexpr int kG  = 1000;
constexpr int kGp = 1008;    // graphs padded to 16

__device__ __forceinline__ unsigned short f2bf(float f) {
  union { float f; unsigned int u; } v; v.f = f;
  unsigned int u = v.u;
  return (unsigned short)((u + 0x7fffu + ((u >> 16) & 1u)) >> 16);
}

// ---- WMMA fragment load ----------------------------------------------------
// 16-bit fragment K-packing (CDNA5 ISA 7.12.2): per half-wave, VGPRs 0..3 hold
// K = base+0..7 (contiguous) and VGPRs 4..7 hold K = base+16..23 (contiguous),
// where base = half*8. So a fragment is exactly two 16-byte loads.
__device__ __forceinline__ v16bf frag_bf(const unsigned short* p, int half) {
  p += half * 8;
  u32x4 lo = *(const u32x4*)(p);        // K base+0..7
  u32x4 hi = *(const u32x4*)(p + 16);   // K base+16..23
  u32x8 w;
  w[0] = lo[0]; w[1] = lo[1]; w[2] = lo[2]; w[3] = lo[3];
  w[4] = hi[0]; w[5] = hi[1]; w[6] = hi[2]; w[7] = hi[3];
  return __builtin_bit_cast(v16bf, w);
}

__device__ __forceinline__ v8f wmma_bf16(v16bf a, v16bf b, v8f c) {
  return __builtin_amdgcn_wmma_f32_16x16x32_bf16(false, a, false, b, (short)0, c,
                                                 false, false);
}

// ---- Tensor Data Mover: 1-D slab load global->LDS --------------------------
#if __has_builtin(__builtin_amdgcn_tensor_load_to_lds)
#define HAVE_TDM 1
// bytes must be a multiple of 8; lds_addr is an LDS byte address.
__device__ __forceinline__ void tdm_load_slab(unsigned lds_addr, const void* gaddr,
                                              unsigned bytes) {
  unsigned elems = bytes >> 3;  // 8-byte elements
  unsigned long long ga = (unsigned long long)gaddr;
  u32x4 g0;
  g0[0] = 1u;                                        // count=1, user descriptor
  g0[1] = lds_addr;                                  // lds_addr
  g0[2] = (unsigned)(ga & 0xffffffffu);              // global_addr[31:0]
  g0[3] = (unsigned)((ga >> 32) & 0x01ffffffu) | 0x80000000u;  // addr[56:32], type=2
  i32x8 g1;
  g1[0] = (3 << 16);                 // data_size = 8B
  g1[1] = (int)((elems & 0xffffu) << 16);            // tensor_dim0[15:0]
  g1[2] = (int)(((elems >> 16) & 0xffffu)) | (1 << 16);  // tensor_dim0[31:16], tensor_dim1=1
  g1[3] = (int)((elems & 0xffffu) << 16);            // tile_dim0 (<=65535 elems)
  g1[4] = 1;                                         // tile_dim1 = 1
  g1[5] = (int)elems;                                // tensor_dim0_stride
  g1[6] = 0;
  g1[7] = 0;
  i32x4 z4 = {};
#if defined(__clang_major__) && __clang_major__ >= 23
  i32x8 z8 = {};
  __builtin_amdgcn_tensor_load_to_lds(g0, g1, z4, z4, z8, 0);
#else
  __builtin_amdgcn_tensor_load_to_lds(g0, g1, z4, z4, 0);
#endif
}
template <int Ncnt>
__device__ __forceinline__ void tdm_wait() {
  __builtin_amdgcn_s_wait_tensorcnt((unsigned short)Ncnt);  // immediate operand
}
#else
#define HAVE_TDM 0
#endif

// ---- Utility kernels -------------------------------------------------------
__global__ void k_zero(float* p, int n) {
  int i = blockIdx.x * 256 + threadIdx.x;
  if (i < n) p[i] = 0.f;
}

__global__ void k_zero16(unsigned short* p, int n) {
  int i = blockIdx.x * 256 + threadIdx.x;
  if (i < n) p[i] = 0;
}

__global__ void k_cvt_bf16(unsigned short* dst, const float* __restrict__ src, int n) {
  int i = blockIdx.x * 256 + threadIdx.x;
  if (i < n) dst[i] = f2bf(src[i]);
}

__global__ void k_copy_pad(float* dst, unsigned short* dstb, const float* __restrict__ src,
                           int valid, int total) {
  int i = blockIdx.x * 256 + threadIdx.x;
  if (i < total) {
    float v = (i < valid) ? src[i] : 0.f;
    dst[i] = v;
    dstb[i] = f2bf(v);
  }
}

// dst[n*K + k] = bf16(src[k*N + n])  (transpose to K-contiguous bf16)
__global__ void k_transpose(unsigned short* dst, const float* __restrict__ src, int K, int N) {
  int i = blockIdx.x * 256 + threadIdx.x;
  if (i >= K * N) return;
  int n = i / K, k = i - n * K;
  dst[i] = f2bf(src[k * N + n]);
}

// ---- Edge MLP layer 1: h_e = relu(edge_attr @ ew1 + eb1), bf16 out ---------
__global__ void k_edge_mlp1(const unsigned short* __restrict__ eab,
                            const unsigned short* __restrict__ ew1t,
                            const float* __restrict__ eb1, unsigned short* __restrict__ he) {
  int tile = blockIdx.x * 8 + (threadIdx.x >> 5);
  if (tile >= (kE / 16) * 8) return;
  int lane = threadIdx.x & 31, idx = lane & 15, half = lane >> 4;
  int tn = tile & 7, tm = tile >> 3;
  v8f acc = {};
#pragma unroll
  for (int ks = 0; ks < 64; ks += 32) {
    v16bf a = frag_bf(eab + (size_t)(tm * 16 + idx) * kC + ks, half);
    v16bf b = frag_bf(ew1t + (size_t)(tn * 16 + idx) * kC + ks, half);
    acc = wmma_bf16(a, b, acc);
  }
  int col = tn * 16 + idx;
  float bias = eb1[col];
#pragma unroll
  for (int v = 0; v < 8; ++v) {
    int row = tm * 16 + half * 8 + v;
    float val = fmaxf(acc[v] + bias, 0.f);
    he[(size_t)row * kHE + col] = f2bf(val);
  }
}

// ---- Fused W_e build + per-edge matvec + scatter ---------------------------
// 10 waves/block, one 16-edge strip per wave (625 blocks * 10 = 6250 strips).
// Per c-channel, the 16KB ew2^T slice is staged into LDS once per block
// (TDM double-buffered when available), then all waves consume it via
// ds_load_b128 fragments. msg[e,h] = sum_c xs[e,c] * ((h_e@ew2)[e,c*64+h]+eb2).
__global__ void __launch_bounds__(320) k_edge_fused(
    const float* __restrict__ x, const unsigned short* __restrict__ he,
    const unsigned short* __restrict__ ew2t, const float* __restrict__ eb2,
    const int* __restrict__ ei, float* __restrict__ agg) {
  __shared__ __align__(16) unsigned short bsh[2][64][128];  // 2 x 16KB B slices
  __shared__ __align__(16) float xsh[10][64][16];           // xs transposed [c][r]
  int wave = threadIdx.x >> 5, lane = threadIdx.x & 31;
  int idx = lane & 15, half = lane >> 4;
  int strip = blockIdx.x * 10 + wave;
  int e0 = strip * 16;

  // Gather source-node features, transposed so per-c reads are contiguous.
  for (int t = lane; t < 16 * 64; t += 32) {
    int r = t >> 6, c = t & 63;
    xsh[wave][c][r] = x[(size_t)ei[e0 + r] * kC + c];
  }

  // Resident A fragments: h_e rows of this strip (K=128 -> 4 ksteps).
  v16bf afr[4];
#pragma unroll
  for (int ks = 0; ks < 4; ++ks)
    afr[ks] = frag_bf(he + (size_t)(e0 + idx) * kHE + ks * 32, half);

  v8f macc[4] = {};  // msg accum: 4 h-tiles x 8 edge-rows per lane

#if HAVE_TDM
  if (wave == 0) tdm_load_slab((unsigned)(unsigned long long)(void*)&bsh[0][0][0],
                               ew2t, 64 * 128 * 2);
#endif

  for (int c = 0; c < 64; ++c) {
#if HAVE_TDM
    int cur = c & 1;
    __syncthreads();  // all waves done reading the buffer we are about to refill
    if (wave == 0) {
      if (c + 1 < 64) {
        tdm_load_slab((unsigned)(unsigned long long)(void*)&bsh[cur ^ 1][0][0],
                      ew2t + (size_t)(c + 1) * 64 * kHE, 64 * 128 * 2);
        tdm_wait<1>();  // slice c landed; next-slice load still in flight
      } else {
        tdm_wait<0>();  // final slice landed
      }
    }
    __syncthreads();
#else
    const int cur = 0;
    __syncthreads();
    {  // cooperative copy of slice c (16KB = 1024 x b128)
      const u32x4* src = (const u32x4*)(ew2t + (size_t)c * 64 * kHE);
      u32x4* dst = (u32x4*)&bsh[0][0][0];
      for (int t = threadIdx.x; t < 1024; t += 320) dst[t] = src[t];
    }
    __syncthreads();
#endif

    // xs for this channel: 8 contiguous rows per lane-half.
    const float4* xp = (const float4*)&xsh[wave][c][half * 8];
    float4 xa = xp[0], xb = xp[1];
    float xv[8] = {xa.x, xa.y, xa.z, xa.w, xb.x, xb.y, xb.z, xb.w};

#pragma unroll
    for (int ht = 0; ht < 4; ++ht) {
      int j = ht * 16 + idx;  // local column within slice
      v8f d = {};
#pragma unroll
      for (int ks = 0; ks < 4; ++ks)
        d = wmma_bf16(afr[ks], frag_bf(&bsh[cur][j][ks * 32], half), d);
      float bias = eb2[c * 64 + j];
#pragma unroll
      for (int v = 0; v < 8; ++v) macc[ht][v] += xv[v] * (d[v] + bias);
    }
  }

#pragma unroll
  for (int ht = 0; ht < 4; ++ht) {
    int h = ht * 16 + idx;
#pragma unroll
    for (int v = 0; v < 8; ++v) {
      int r = half * 8 + v;
      atomicAdd(&agg[(size_t)ei[kE + e0 + r] * kH + h], macc[ht][v]);
    }
  }
}

// ---- x@root + agg + bias, relu, BN-stat accumulation -----------------------
__global__ void k_xroot(const unsigned short* __restrict__ xb,
                        const unsigned short* __restrict__ rootT,
                        const float* __restrict__ agg, const float* __restrict__ bias,
                        float* __restrict__ y, float* __restrict__ stats) {
  const int nT = (kNp / 16) * 4;
  int tile = blockIdx.x * 8 + (threadIdx.x >> 5);
  if (tile >= nT) return;
  int lane = threadIdx.x & 31, idx = lane & 15, half = lane >> 4;
  int tn = tile & 3, tm = tile >> 2;
  v8f acc = {};
#pragma unroll
  for (int ks = 0; ks < 64; ks += 32) {
    v16bf a = frag_bf(xb + (size_t)(tm * 16 + idx) * kC + ks, half);
    v16bf b = frag_bf(rootT + (size_t)(tn * 16 + idx) * kC + ks, half);
    acc = wmma_bf16(a, b, acc);
  }
  int col = tn * 16 + idx;
  float bb = bias[col];
  float s = 0.f, s2 = 0.f;
#pragma unroll
  for (int v = 0; v < 8; ++v) {
    int row = tm * 16 + half * 8 + v;
    if (row < kN) {
      float val = fmaxf(acc[v] + agg[(size_t)row * kH + col] + bb, 0.f);
      y[(size_t)row * kH + col] = val;
      s += val;
      s2 += val * val;
    }
  }
  atomicAdd(&stats[col], s);
  atomicAdd(&stats[64 + col], s2);
}

__global__ void k_bn_finalize(float* stats, const float* __restrict__ gamma,
                              const float* __restrict__ beta) {
  int h = threadIdx.x;  // 64 threads
  float inv = 1.f / (float)kN;
  float mu = stats[h] * inv;
  float var = stats[64 + h] * inv - mu * mu;
  float sc = gamma[h] * rsqrtf(var + 1e-5f);
  stats[128 + h] = sc;
  stats[192 + h] = beta[h] - mu * sc;
}

__global__ void k_normalize(float* __restrict__ xn, unsigned short* __restrict__ xnb,
                            const float* __restrict__ y, const float* __restrict__ stats) {
  int i = blockIdx.x * 256 + threadIdx.x;
  if (i >= kN * kH) return;
  int h = i & 63;
  float v = y[i] * stats[128 + h] + stats[192 + h];
  xn[i] = v;
  xnb[i] = f2bf(v);
}

__global__ void k_pool(float* __restrict__ pooled, const float* __restrict__ x,
                       const int* __restrict__ batch) {
  int i = blockIdx.x * 256 + threadIdx.x;
  if (i >= kN * kH) return;
  int node = i >> 6, h = i & 63;
  atomicAdd(&pooled[(size_t)batch[node] * kH + h], x[i]);
}

// ---- Readout: pooled[1008,64] @ ro_w[64,256] + ro_b ------------------------
__global__ void k_readout(float* __restrict__ out, const unsigned short* __restrict__ pb,
                          const unsigned short* __restrict__ rowt,
                          const float* __restrict__ rob) {
  int tile = blockIdx.x * 8 + (threadIdx.x >> 5);  // 63*16 = 1008 tiles
  if (tile >= (kGp / 16) * 16) return;
  int lane = threadIdx.x & 31, idx = lane & 15, half = lane >> 4;
  int tn = tile & 15, tm = tile >> 4;
  v8f acc = {};
#pragma unroll
  for (int ks = 0; ks < 64; ks += 32) {
    v16bf a = frag_bf(pb + (size_t)(tm * 16 + idx) * kH + ks, half);
    v16bf b = frag_bf(rowt + (size_t)(tn * 16 + idx) * kH + ks, half);
    acc = wmma_bf16(a, b, acc);
  }
  int col = tn * 16 + idx;
  float rb = rob[col];
#pragma unroll
  for (int v = 0; v < 8; ++v) {
    int row = tm * 16 + half * 8 + v;
    if (row < kG) out[(size_t)row * kD + col] = acc[v] + rb;
  }
}

// ---- Host launch -----------------------------------------------------------
extern "C" void kernel_launch(void* const* d_in, const int* in_sizes, int n_in,
                              void* d_out, int out_size, void* d_ws, size_t ws_size,
                              hipStream_t stream) {
  (void)in_sizes; (void)n_in; (void)out_size; (void)ws_size;
  const float* x_in   = (const float*)d_in[0];
  const int*   ei     = (const int*)d_in[1];
  const float* ea     = (const float*)d_in[2];
  const int*   batch  = (const int*)d_in[3];
  const float* ew1    = (const float*)d_in[4];
  const float* eb1    = (const float*)d_in[5];
  const float* ew2    = (const float*)d_in[6];
  const float* eb2    = (const float*)d_in[7];
  const float* roots  = (const float*)d_in[8];
  const float* biases = (const float*)d_in[9];
  const float* gamma  = (const float*)d_in[10];
  const float* beta   = (const float*)d_in[11];
  const float* ro_w   = (const float*)d_in[12];
  const float* ro_b   = (const float*)d_in[13];
  float* out = (float*)d_out;

  char* ws = (char*)d_ws;
  auto alloc = [&](size_t bytes) { char* p = ws; ws += (bytes + 255) & ~(size_t)255; return p; };
  unsigned short* he     = (unsigned short*)alloc((size_t)kE * kHE * 2);
  unsigned short* eab    = (unsigned short*)alloc((size_t)kE * kC * 2);
  unsigned short* ew1t   = (unsigned short*)alloc((size_t)kHE * kC * 2);
  unsigned short* ew2t   = (unsigned short*)alloc((size_t)(kC * kH) * kHE * 2);
  unsigned short* rootst = (unsigned short*)alloc((size_t)3 * kC * kH * 2);
  unsigned short* rowt   = (unsigned short*)alloc((size_t)kD * kH * 2);
  unsigned short* xbf0   = (unsigned short*)alloc((size_t)kNp * kH * 2);
  unsigned short* xbf1   = (unsigned short*)alloc((size_t)kNp * kH * 2);
  unsigned short* plb    = (unsigned short*)alloc((size_t)kGp * kH * 2);
  float* xbuf0  = (float*)alloc((size_t)kNp * kH * 4);
  float* xbuf1  = (float*)alloc((size_t)kNp * kH * 4);
  float* ybuf   = (float*)alloc((size_t)kNp * kH * 4);
  float* agg    = (float*)alloc((size_t)kN * kH * 4);
  float* pooled = (float*)alloc((size_t)kGp * kH * 4);
  float* stats  = (float*)alloc(256 * 4);

  // Weights -> K-contiguous bf16; activations -> bf16 mirrors.
  k_transpose<<<(kC * kHE + 255) / 256, 256, 0, stream>>>(ew1t, ew1, kC, kHE);
  k_transpose<<<(kHE * kC * kH + 255) / 256, 256, 0, stream>>>(ew2t, ew2, kHE, kC * kH);
  for (int l = 0; l < 3; ++l)
    k_transpose<<<(kC * kH + 255) / 256, 256, 0, stream>>>(rootst + l * kC * kH,
                                                           roots + l * kC * kH, kC, kH);
  k_transpose<<<(kH * kD + 255) / 256, 256, 0, stream>>>(rowt, ro_w, kH, kD);
  k_cvt_bf16<<<(kE * kC + 255) / 256, 256, 0, stream>>>(eab, ea, kE * kC);
  k_copy_pad<<<(kNp * kH + 255) / 256, 256, 0, stream>>>(xbuf0, xbf0, x_in, kN * kH, kNp * kH);
  k_zero<<<(kNp * kH + 255) / 256, 256, 0, stream>>>(xbuf1, kNp * kH);
  k_zero16<<<(kNp * kH + 255) / 256, 256, 0, stream>>>(xbf1, kNp * kH);

  // Shared edge MLP layer 1 (computed once).
  k_edge_mlp1<<<((kE / 16) * 8 + 7) / 8, 256, 0, stream>>>(eab, ew1t, eb1, he);

  for (int l = 0; l < 3; ++l) {
    const float*          xc  = (l & 1) ? xbuf1 : xbuf0;
    const unsigned short* xcb = (l & 1) ? xbf1 : xbf0;
    float*          xn  = (l & 1) ? xbuf0 : xbuf1;
    unsigned short* xnb = (l & 1) ? xbf0 : xbf1;
    k_zero<<<(kN * kH + 255) / 256, 256, 0, stream>>>(agg, kN * kH);
    k_zero<<<1, 256, 0, stream>>>(stats, 256);
    k_edge_fused<<<kE / 16 / 10, 320, 0, stream>>>(xc, he, ew2t, eb2, ei, agg);
    k_xroot<<<((kNp / 16) * 4 + 7) / 8, 256, 0, stream>>>(xcb, rootst + l * kC * kH, agg,
                                                          biases + l * kH, ybuf, stats);
    k_bn_finalize<<<1, 64, 0, stream>>>(stats, gamma, beta);
    k_normalize<<<(kN * kH + 255) / 256, 256, 0, stream>>>(xn, xnb, ybuf, stats);
  }

  // Global add pool + readout (final node features live in xbuf1 after L=3).
  k_zero<<<(kGp * kH + 255) / 256, 256, 0, stream>>>(pooled, kGp * kH);
  k_pool<<<(kN * kH + 255) / 256, 256, 0, stream>>>(pooled, xbuf1, batch);
  k_cvt_bf16<<<(kGp * kH + 255) / 256, 256, 0, stream>>>(plb, pooled, kGp * kH);
  k_readout<<<((kGp / 16) * 16) / 8, 256, 0, stream>>>(out, plb, rowt, ro_b);
}
